// T2L2Loss_3693671874704
// MI455X (gfx1250) — compile-verified, matrix operand order checked by reading, not run on hardware
//
#include <hip/hip_runtime.h>
#include <hip/hip_bf16.h>
#include <stdint.h>

#define B_ROWS  1024
#define D_DIM   16384
#define K_POS   8
#define NEG_N   48          // 6*k
#define THREADS 256
#define PER_T   (D_DIM / THREADS)   // 64 elements per thread
#define N_TOTAL (B_ROWS * D_DIM)    // 2^24
#define N_HALF  (N_TOTAL / 2)       // 2^23

typedef __attribute__((ext_vector_type(2)))  float        v2f;
typedef __attribute__((ext_vector_type(8)))  float        v8f;
typedef __attribute__((ext_vector_type(16))) _Float16     v16h;
typedef __attribute__((ext_vector_type(4)))  unsigned int v4u;
typedef __attribute__((ext_vector_type(8)))  int          v8i;
typedef __attribute__((ext_vector_type(4)))  int          v4i;

// softplus(x) = log(1+e^x), numerically stable
__device__ __forceinline__ float softplus_f(float x) {
  return fmaxf(x, 0.0f) + log1pf(expf(-fabsf(x)));
}

__device__ __forceinline__ void tf_round(uint32_t& x0, uint32_t& x1, int r) {
  x0 += x1;
  x1 = (x1 << r) | (x1 >> (32 - r));
  x1 ^= x0;
}

// Threefry-2x32, 20 rounds, matching JAX's implementation.
__device__ __forceinline__ uint2 threefry2x32(uint32_t k0, uint32_t k1,
                                              uint32_t x0, uint32_t x1) {
  uint32_t ks2 = 0x1BD11BDAu ^ k0 ^ k1;
  x0 += k0; x1 += k1;
  tf_round(x0, x1, 13); tf_round(x0, x1, 15); tf_round(x0, x1, 26); tf_round(x0, x1, 6);
  x0 += k1; x1 += ks2 + 1u;
  tf_round(x0, x1, 17); tf_round(x0, x1, 29); tf_round(x0, x1, 16); tf_round(x0, x1, 24);
  x0 += ks2; x1 += k0 + 2u;
  tf_round(x0, x1, 13); tf_round(x0, x1, 15); tf_round(x0, x1, 26); tf_round(x0, x1, 6);
  x0 += k0; x1 += k1 + 3u;
  tf_round(x0, x1, 17); tf_round(x0, x1, 29); tf_round(x0, x1, 16); tf_round(x0, x1, 24);
  x0 += k1; x1 += ks2 + 4u;
  tf_round(x0, x1, 13); tf_round(x0, x1, 15); tf_round(x0, x1, 26); tf_round(x0, x1, 6);
  x0 += ks2; x1 += k0 + 5u;
  return make_uint2(x0, x1);
}

// jax.random.uniform(key(42), (B,D)) element at flat index g.
// JAX splits iota(n) into two counter halves; element g<n/2 takes out0 of
// (g, g+n/2), else out1 of (g-n/2, g). bits -> [0,1) via mantissa trick.
__device__ __forceinline__ float jax_uniform_at(uint32_t g) {
  uint32_t bits;
  if (g < (uint32_t)N_HALF) bits = threefry2x32(0u, 42u, g, g + (uint32_t)N_HALF).x;
  else                      bits = threefry2x32(0u, 42u, g - (uint32_t)N_HALF, g).y;
  return __uint_as_float((bits >> 9) | 0x3F800000u) - 1.0f;
}

// ---------------------------------------------------------------------------
// Kernel 1: one workgroup per row. Stages target row into LDS with the TDM,
// regenerates JAX noise, bisects the top-48 threshold, accumulates the BCE
// partial loss for the row.
// LDS layout (dynamic, base offset 0):
//   [0,              64KB) : target row (TDM destination)
//   [64KB,          128KB) : noise row
//   [128KB, 128KB + 64B  ) : count / threshold / per-wave partials
// ---------------------------------------------------------------------------
__global__ void row_loss_kernel(const float* __restrict__ pred,
                                const float* __restrict__ target,
                                float* __restrict__ rowloss) {
  extern __shared__ char smem[];
  float* tg   = (float*)smem;                       // D floats
  float* nz   = (float*)(smem + D_DIM * 4);         // D floats
  int*   scnt = (int*)  (smem + 2 * D_DIM * 4);
  float* sthr = (float*)(smem + 2 * D_DIM * 4 + 4);
  float* wsum = (float*)(smem + 2 * D_DIM * 4 + 8); // 8 floats

  const int row = blockIdx.x;
  const int tid = threadIdx.x;
  const float* predrow = pred   + (size_t)row * D_DIM;
  const float* tgtrow  = target + (size_t)row * D_DIM;

  __builtin_prefetch(predrow, 0, 1);   // global_prefetch_b8 into L2

#if __has_builtin(__builtin_amdgcn_tensor_load_to_lds)
  // --- TDM: DMA the 16384x1 f32 target row into LDS offset 0 ---
  if (tid < 32) {   // single wave issues the tensor op (EXEC-independent)
    uint64_t ga = (uint64_t)(uintptr_t)tgtrow;
    v4u g0;
    g0[0] = 1u;                                        // count=1, gather off
    g0[1] = 0u;                                        // lds_addr = 0
    g0[2] = (uint32_t)(ga & 0xFFFFFFFFu);              // global_addr lo
    g0[3] = (uint32_t)((ga >> 32) & 0x1FFFFFFu)        // global_addr hi
          | (2u << 30);                                // type = 2 ("image")
    v8i g1;
    g1[0] = (int)(2u << 16);                           // data_size = 4B
    g1[1] = (int)((uint32_t)D_DIM << 16);              // tensor_dim0[15:0] @bits63:48
    g1[2] = (int)(((uint32_t)D_DIM >> 16) | (1u << 16)); // dim0 hi | tensor_dim1=1
    g1[3] = (int)((uint32_t)D_DIM << 16);              // tile_dim0 @bits127:112
    g1[4] = 1;                                         // tile_dim1=1, tile_dim2=0
    g1[5] = D_DIM;                                     // tensor_dim0_stride lo
    g1[6] = 0;
    g1[7] = 0;
    v4i gz4 = {0, 0, 0, 0};                            // 2D tensor: groups 2/3 unused
    v8i gz8 = {0, 0, 0, 0, 0, 0, 0, 0};                // unused trailing group
    // clang-23 / therock-10.0 6-arg form:
    // (uint32x4 g0, int32x8 g1, int32x4, int32x4, int32x8, i32 cpol)
    __builtin_amdgcn_tensor_load_to_lds(g0, g1, gz4, gz4, gz8, 0);
    __builtin_amdgcn_s_wait_tensorcnt(0);
  }
  __syncthreads();
#else
  for (int j = tid; j < D_DIM; j += THREADS) tg[j] = tgtrow[j];
  __syncthreads();
#endif

  float loss = 0.0f;

  // Pass 1: regenerate noise, mask positives, accumulate positive BCE terms.
  for (int i = 0; i < PER_T; ++i) {
    int j = i * THREADS + tid;                 // interleaved: conflict-free LDS
    float u = jax_uniform_at((uint32_t)row * D_DIM + (uint32_t)j);
    if (tg[j] > 0.0f) {
      u = -1.0f;                               // exclude positive from sampling
      loss += softplus_f(-predrow[j]);         // y=1 term
    }
    nz[j] = u;
  }
  if (tid == 0) *sthr = -1.0f;
  __syncthreads();

  // Pass 2: bisect threshold t with count(noise > t) == 48 (top-48 membership).
  float lo = 0.0f, hi = 1.0f;
  for (int it = 0; it < 28; ++it) {
    float mid = 0.5f * (lo + hi);
    if (tid == 0) *scnt = 0;
    __syncthreads();
    int c = 0;
    for (int i = 0; i < PER_T; ++i) c += (nz[i * THREADS + tid] > mid) ? 1 : 0;
    atomicAdd(scnt, c);                        // LDS ds_add
    __syncthreads();
    int total = *scnt;                         // uniform across block
    if (total == NEG_N && tid == 0) *sthr = mid;
    if (total > NEG_N) lo = mid; else hi = mid;
    __syncthreads();
  }
  float thr = *sthr;

  // Pass 3: accumulate negative BCE terms.
  for (int i = 0; i < PER_T; ++i) {
    int j = i * THREADS + tid;
    if (nz[j] > thr) loss += softplus_f(predrow[j]);  // y=0 term
  }

  // wave32 reduction, then cross-wave via LDS
  for (int off = 16; off > 0; off >>= 1) loss += __shfl_down(loss, off, 32);
  if ((tid & 31) == 0) wsum[tid >> 5] = loss;
  __syncthreads();
  if (tid == 0) {
    float s = 0.0f;
    for (int w = 0; w < THREADS / 32; ++w) s += wsum[w];
    rowloss[row] = s;
  }
}

// ---------------------------------------------------------------------------
// Kernel 2: reduce the 1024 per-row losses to a scalar with WMMA.
// D = A(16x4 partials) x B(4x16 ones) + C  accumulates exact f32 row sums.
// Single full wave32 (EXEC all ones, as WMMA requires).
// ---------------------------------------------------------------------------
__global__ void wmma_reduce_kernel(const float* __restrict__ rowloss,
                                   float* __restrict__ out) {
  const int lane = threadIdx.x;      // 0..31
  const int m = lane & 15;
  v8f c = {};
#if __has_builtin(__builtin_amdgcn_wmma_f32_16x16x4_f32)
  // A 16x4 f32 layout: lanes 0-15 hold (K0,K1), lanes 16-31 hold (K2,K3), M=lane&15
  const int kb = (lane >> 4) << 1;   // 0 or 2
  v2f bones; bones.x = 1.0f; bones.y = 1.0f;
  for (int it = 0; it < B_ROWS / 64; ++it) {       // 16 WMMAs
    const float* chunk = rowloss + it * 64;        // A[m][k] = chunk[m*4+k]
    v2f a;
    a.x = chunk[m * 4 + kb];
    a.y = chunk[m * 4 + kb + 1];
    c = __builtin_amdgcn_wmma_f32_16x16x4_f32(false, a, false, bones,
                                              (short)0, c, false, false);
  }
#else
  // Fallback: probe-confirmed f16 WMMA (B = ones so layout-independent sum).
  v16h bones;
  for (int i = 0; i < 16; ++i) bones[i] = (_Float16)1.0f;
  const int kb = (lane >> 4) * 8;    // 0 or 8
  for (int it = 0; it < B_ROWS / 512; ++it) {      // 2 WMMAs
    const float* chunk = rowloss + it * 512;       // A[m][k] = chunk[m*32+k]
    v16h a;
    for (int i = 0; i < 8; ++i) a[i]     = (_Float16)chunk[m * 32 + kb + i];
    for (int i = 0; i < 8; ++i) a[8 + i] = (_Float16)chunk[m * 32 + 16 + kb + i];
    c = __builtin_amdgcn_wmma_f32_16x16x32_f16(false, a, false, bones,
                                               (short)0, c, false, false);
  }
#endif
  // C[m][n]: every column identical. lane0 holds M=0..7 (N=0), lane16 M=8..15.
  float s = 0.0f;
  for (int i = 0; i < 8; ++i) s += c[i];
  float other = __shfl(s, lane ^ 16, 32);
  if (lane == 0) out[0] = s + other;
}

extern "C" void kernel_launch(void* const* d_in, const int* in_sizes, int n_in,
                              void* d_out, int out_size, void* d_ws, size_t ws_size,
                              hipStream_t stream) {
  const float* pred   = (const float*)d_in[0];
  const float* target = (const float*)d_in[1];
  // d_in[2] is k == 8, baked into the kernel constants.
  float* rowloss = (float*)d_ws;     // 1024 floats of scratch

  const size_t shmem = (size_t)2 * D_DIM * sizeof(float) + 64;  // ~128KB (<320KB/WGP)
  (void)hipFuncSetAttribute((const void*)row_loss_kernel,
                            hipFuncAttributeMaxDynamicSharedMemorySize,
                            (int)shmem);

  row_loss_kernel<<<B_ROWS, THREADS, shmem, stream>>>(pred, target, rowloss);
  wmma_reduce_kernel<<<1, 32, 0, stream>>>(rowloss, (float*)d_out);
}
